// RGLRU_27401891349130
// MI455X (gfx1250) — matrix operand order repaired and checked by reading, not compiled, not run on hardware
//
#include <hip/hip_runtime.h>

// RG-LRU for gfx1250: bf16 WMMA gate projections + chunked parallel scan.
// B=8, T=4096, D=1024, H=8, bw=128.

typedef __attribute__((ext_vector_type(16))) __bf16 v16bf;
typedef __attribute__((ext_vector_type(8)))  float  v8f;

#define Bc   8
#define Tc   4096
#define Dc   1024
#define Hc   8
#define BWc  128
#define NCc  64           // scan chunks along T
#define Lc   (Tc / NCc)   // 64 steps per chunk
#define CHNc (Bc * Dc)    // 8192 independent channels

union BF16x16 { unsigned int u[8]; v16bf v; };

__device__ __forceinline__ unsigned short f2bf(float f) {
  unsigned int u = __float_as_uint(f);
  u += 0x7fffu + ((u >> 16) & 1u);        // round-to-nearest-even
  return (unsigned short)(u >> 16);
}
__device__ __forceinline__ float sigmoidf_(float v) { return 1.0f / (1.0f + __expf(-v)); }
__device__ __forceinline__ float softplusf_(float v) {
  return (v > 20.0f) ? v : __logf(1.0f + __expf(v));
}

// ---------------------------------------------------------------------------
// Kernel 1: block-diagonal gate GEMMs via v_wmma_f32_16x16x32_bf16, fused
// with the pointwise gate math. One block = (b, h, 128-row tile of T).
// 8 waves x 16 rows each; each wave issues 64 WMMAs.
// Writes: normed_x -> y region of d_out, per-element decay a -> workspace.
// ---------------------------------------------------------------------------
__global__ __launch_bounds__(256)
void rglru_gates_wmma(const float* __restrict__ x,
                      const float* __restrict__ a_param,
                      const float* __restrict__ ig_w, const float* __restrict__ ig_b,
                      const float* __restrict__ ag_w, const float* __restrict__ ag_b,
                      float* __restrict__ nx_out,   // normed_x (y region of d_out)
                      float* __restrict__ a_out) {  // decay a (workspace)
  // Both gate weight blocks in LDS as bf16, stored transposed [j][i] so the
  // K-pairs of the WMMA B-matrix layout are contiguous 32-bit reads. 64 KB.
  __shared__ unsigned short sW[2][BWc * BWc];

  const int tid    = threadIdx.x;
  const int tilesT = Tc / 128;
  const int tileT  = blockIdx.x % tilesT;
  const int bh     = blockIdx.x / tilesT;
  const int h      = bh % Hc;
  const int b      = bh / Hc;
  const int t0     = tileT * 128;

  for (int idx = tid; idx < BWc * BWc; idx += 256) {
    const int i = idx / BWc;               // input dim (K)
    const int j = idx % BWc;               // output dim (N)
    const int src = (h * BWc + i) * BWc + j;
    sW[0][j * BWc + i] = f2bf(ig_w[src]);
    sW[1][j * BWc + i] = f2bf(ag_w[src]);
  }
  __syncthreads();

  const int lane = tid & 31;
  const int wave = tid >> 5;
  const int hi   = lane >> 4;              // half-wave select
  const int lc   = lane & 15;

  // A fragments (16x32 bf16 each, ISA layout): lanes 0-15 and 16-31 both hold
  // M=0..15; per-VGPR K = 16*(v/4) + 8*hi + 2*(v%4). Built from global f32.
  const int    arow = t0 + wave * 16 + lc;
  const float* xrow = x + ((size_t)b * Tc + arow) * Dc + h * BWc;
  v16bf aF[4];
#pragma unroll
  for (int ks = 0; ks < 4; ++ks) {
    BF16x16 u;
#pragma unroll
    for (int v = 0; v < 8; ++v) {
      const int k = ks * 32 + (v >> 2) * 16 + hi * 8 + (v & 3) * 2;
      const float2 p = *reinterpret_cast<const float2*>(xrow + k);
      u.u[v] = (unsigned int)f2bf(p.x) | ((unsigned int)f2bf(p.y) << 16);
    }
    aF[ks] = u.v;
  }

  // B fragment (32x16 bf16): lane holds column N = lane%16; per-VGPR
  // K = 16*(v/4) + 8*hi + 2*(v%4), two K per VGPR (matches A).
  auto loadB = [&](int g, int nt, int ks) -> v16bf {
    BF16x16 u;
    const int j = nt * 16 + lc;
#pragma unroll
    for (int v = 0; v < 8; ++v) {
      const int k = ks * 32 + (v >> 2) * 16 + hi * 8 + (v & 3) * 2;
      u.u[v] = *reinterpret_cast<const unsigned int*>(&sW[g][j * BWc + k]);
    }
    return u.v;
  };

#pragma unroll
  for (int nt = 0; nt < 8; ++nt) {
    v8f ci = {};
    v8f ca = {};
#pragma unroll
    for (int ks = 0; ks < 4; ++ks)
      ci = __builtin_amdgcn_wmma_f32_16x16x32_bf16(false, aF[ks], false, loadB(0, nt, ks),
                                                   (short)0, ci, false, false);
#pragma unroll
    for (int ks = 0; ks < 4; ++ks)
      ca = __builtin_amdgcn_wmma_f32_16x16x32_bf16(false, aF[ks], false, loadB(1, nt, ks),
                                                   (short)0, ca, false, false);

    // Fused pointwise epilogue. C/D layout: VGPR r -> row r + 8*hi, col lane%16.
    const int   j    = nt * 16 + lc;
    const int   dcol = h * BWc + j;
    const float sp   = softplusf_(a_param[dcol]);
    const float bi   = ig_b[h * BWc + j];
    const float ba   = ag_b[h * BWc + j];
#pragma unroll
    for (int r = 0; r < 8; ++r) {
      const int    t  = t0 + wave * 16 + hi * 8 + r;
      const size_t gi = ((size_t)b * Tc + t) * Dc + dcol;
      const float xv = x[gi];
      const float gx = sigmoidf_(ci[r] + bi);
      const float ga = sigmoidf_(ca[r] + ba);
      const float la = -8.0f * ga * sp;
      const float av = __expf(la);
      const float a2 = __expf(2.0f * la);
      const float mu = sqrtf(fmaxf(1.0f - a2, 1e-7f));
      nx_out[gi] = xv * gx * mu;
      a_out[gi]  = av;
    }
  }
}

// ---------------------------------------------------------------------------
// Scan (h_t = a_{t-1} * h_{t-1} + nx_t, matching the reference's shifted
// closed form; ash_0 := 0). Chunked: local scans, carries, fix-up.
// ---------------------------------------------------------------------------
__global__ __launch_bounds__(256)
void rglru_scan_chunks(const float* __restrict__ a_arr,
                       float* __restrict__ y,            // in: nx, out: local h
                       float* __restrict__ chunkA,       // [NCc][CHNc]
                       float* __restrict__ chunkH) {     // [NCc][CHNc]
  const int gid   = blockIdx.x * blockDim.x + threadIdx.x;
  const int ch    = gid & (CHNc - 1);
  const int chunk = gid >> 13;                           // CHNc == 8192 == 2^13
  const int d = ch % Dc, b = ch / Dc;
  const size_t base = (size_t)b * Tc * Dc + d;

  float hcur = 0.0f, Ap = 1.0f;
  const int tstart = chunk * Lc;
  for (int s = 0; s < Lc; ++s) {
    const int   t   = tstart + s;
    const float ash = (t == 0) ? 0.0f : a_arr[base + (size_t)(t - 1) * Dc];
    const float nx  = y[base + (size_t)t * Dc];
    hcur = ash * hcur + nx;
    Ap  *= ash;
    y[base + (size_t)t * Dc] = hcur;
  }
  chunkA[chunk * CHNc + ch] = Ap;
  chunkH[chunk * CHNc + ch] = hcur;
}

__global__ __launch_bounds__(256)
void rglru_scan_carries(const float* __restrict__ chunkA,
                        const float* __restrict__ chunkH,
                        float* __restrict__ carry) {     // [NCc][CHNc]
  const int ch = blockIdx.x * blockDim.x + threadIdx.x;  // CHNc threads
  float c = 0.0f;
  for (int k = 0; k < NCc; ++k) {
    carry[k * CHNc + ch] = c;
    c = chunkH[k * CHNc + ch] + chunkA[k * CHNc + ch] * c;
  }
}

__global__ __launch_bounds__(256)
void rglru_scan_fix(const float* __restrict__ a_arr,
                    const float* __restrict__ carry,
                    float* __restrict__ y,
                    float* __restrict__ last_h) {        // [B*D]
  const int gid   = blockIdx.x * blockDim.x + threadIdx.x;
  const int ch    = gid & (CHNc - 1);
  const int chunk = gid >> 13;
  const int d = ch % Dc, b = ch / Dc;
  const size_t base = (size_t)b * Tc * Dc + d;

  const float c = carry[chunk * CHNc + ch];
  float p = 1.0f;
  const int tstart = chunk * Lc;
  for (int s = 0; s < Lc; ++s) {
    const int   t   = tstart + s;
    const float ash = (t == 0) ? 0.0f : a_arr[base + (size_t)(t - 1) * Dc];
    p *= ash;
    const float hv = y[base + (size_t)t * Dc] + p * c;
    y[base + (size_t)t * Dc] = hv;
    if (t == Tc - 1) last_h[ch] = hv;
  }
}

// ---------------------------------------------------------------------------
extern "C" void kernel_launch(void* const* d_in, const int* in_sizes, int n_in,
                              void* d_out, int out_size, void* d_ws, size_t ws_size,
                              hipStream_t stream) {
  const float* x   = (const float*)d_in[0];
  // d_in[1] = segment_pos (unused by the reference math)
  const float* ap  = (const float*)d_in[2];
  const float* igw = (const float*)d_in[3];
  const float* igb = (const float*)d_in[4];
  const float* agw = (const float*)d_in[5];
  const float* agb = (const float*)d_in[6];

  float* y    = (float*)d_out;                     // [B,T,D]
  float* last = y + (size_t)Bc * Tc * Dc;          // [B,D]

  float* wsA      = (float*)d_ws;                  // [B,T,D] decay a  (~134 MB)
  float* wsAprod  = wsA + (size_t)Bc * Tc * Dc;    // [NCc][CHNc]
  float* wsHend   = wsAprod + (size_t)NCc * CHNc;  // [NCc][CHNc]
  float* wsCarry  = wsHend + (size_t)NCc * CHNc;   // [NCc][CHNc]

  rglru_gates_wmma<<<Bc * Hc * (Tc / 128), 256, 0, stream>>>(
      x, ap, igw, igb, agw, agb, y, wsA);

  rglru_scan_chunks<<<(CHNc * NCc) / 256, 256, 0, stream>>>(wsA, y, wsAprod, wsHend);

  rglru_scan_carries<<<CHNc / 256, 256, 0, stream>>>(wsAprod, wsHend, wsCarry);

  rglru_scan_fix<<<(CHNc * NCc) / 256, 256, 0, stream>>>(wsA, wsCarry, y, last);
}
